// FreeYOLO_34059090657489
// MI455X (gfx1250) — matrix operand order repaired and checked by reading, not compiled, not run on hardware
//
#include <hip/hip_runtime.h>
#include <hip/hip_bf16.h>

#define NCCH    80
#define NOUT    85
#define NPAD    96
#define TOPKN   1000
#define NCAND   3000
#define NSORT   4096
#define NWORDS  94          // ceil(3000/32)
#define CONF_T  0.05f
#define NMS_THR 0.6f
#define IMG_SZ  2048.0f
#define CLSOFF  100000.0f

typedef __attribute__((ext_vector_type(16))) __bf16 v16bf;
typedef __attribute__((ext_vector_type(8)))  float  v8f;

__device__ __forceinline__ float sigm(float x){ return 1.0f/(1.0f + __expf(-x)); }

// ---------------------------------------------------------------------------
// Zero all mutable control state every call (deterministic under graph replay)
// ---------------------------------------------------------------------------
__global__ void init_kernel(unsigned* hist, unsigned* cnt, float* cscore,
                            int* clabel, float* cbox, unsigned* validw, float* keep)
{
    int i = blockIdx.x * blockDim.x + threadIdx.x;   // 12*256 = 3072 threads
    hist[i] = 0u;                                    // exactly 3*1024 bins
    if (i < 3)    cnt[i] = 0u;
    if (i < 128)  validw[i] = 0u;
    if (i < NCAND){
        cscore[i] = 0.0f; clabel[i] = 0; keep[i] = 0.0f;
        float4 z = make_float4(0.f,0.f,0.f,0.f);
        *(float4*)(cbox + 4*(size_t)i) = z;
    }
}

// ---------------------------------------------------------------------------
// Pack f32 weights into bf16 WMMA B-fragments: [kk][nt][lane][e]
// B (32x16): lane L -> column N=L&15 ; K = (L>=16 ? 8:0) + (e&7) + (e>=8 ? 16:0)
// ---------------------------------------------------------------------------
__global__ void pack_weights_kernel(const float* __restrict__ w0, const float* __restrict__ w1,
                                    const float* __restrict__ w2, __bf16* __restrict__ wpack)
{
    int idx = blockIdx.x * blockDim.x + threadIdx.x;
    if (idx >= 172032) return;                       // (256+512+1024)*96
    int base, K; const float* w;
    if (idx < 24576)      { base = 0;     K = 256;  w = w0; }
    else if (idx < 73728) { base = 24576; K = 512;  w = w1; }
    else                  { base = 73728; K = 1024; w = w2; }
    int i2   = idx - base;
    int e    = i2 & 15;
    int lane = (i2 >> 4) & 31;
    int rest = i2 >> 9;
    int nt   = rest % 6;
    int kk   = rest / 6;
    int k = kk*32 + (lane >> 4)*8 + (e & 7) + ((e & 8) ? 16 : 0);
    int n = nt*16 + (lane & 15);
    float v = (n < NOUT) ? w[n*K + k] : 0.0f;
    wpack[idx] = (__bf16)v;
}

__global__ void pack_bias_kernel(const float* b0, const float* b1, const float* b2, float* biasp)
{
    int i = blockIdx.x * blockDim.x + threadIdx.x;
    if (i >= 3*NPAD) return;
    int lvl = i / NPAD, n = i % NPAD;
    const float* b = (lvl == 0) ? b0 : (lvl == 1) ? b1 : b2;
    biasp[i] = (n < NOUT) ? b[n] : 0.0f;
}

// ---------------------------------------------------------------------------
// Detection head: bf16 WMMA GEMM (16 positions x 96 outputs per wave) + post.
// KSTEPS=C/32, H, STRD are compile-time: all feature offsets k*H*H become
// constants (immediate-offset loads), K loop software-pipelines.
// ---------------------------------------------------------------------------
template<int KSTEPS, int H, int STRD>
__global__ void __launch_bounds__(128)
head_kernel(const float* __restrict__ feat, const __bf16* __restrict__ wpack,
            const float* __restrict__ biasp, float* __restrict__ scores,
            int* __restrict__ labels, float* __restrict__ boxes)
{
    __shared__ float lds[4][16][NPAD];               // 24 KB
    constexpr int HW = H * H;
    const int lane  = threadIdx.x & 31;
    const int wave  = threadIdx.x >> 5;
    const int tile  = blockIdx.x * 4 + wave;
    const int p0    = tile * 16;
    const int m     = lane & 15;
    const int khalf = lane >> 4;
    const float* fp = feat + p0 + m + (size_t)khalf * 8 * HW;  // lane's K base

    v8f acc[6];
    #pragma unroll
    for (int nt = 0; nt < 6; ++nt){
        v8f z = {0.f,0.f,0.f,0.f,0.f,0.f,0.f,0.f};
        acc[nt] = z;
    }

    #pragma unroll 2
    for (int kk = 0; kk < KSTEPS; ++kk){
        // speculative prefetch of next K-step's feature lines (faults dropped)
        __builtin_prefetch(fp + (size_t)(kk + 1) * 32 * HW, 0, 1);

        // Preload ALL six B fragments so the loads clause together and drain
        // while earlier WMMAs execute in the matrix pipe.
        const __bf16* wb = wpack + ((size_t)(kk*6)*32 + lane)*16;
        v16bf bfr[6];
        #pragma unroll
        for (int nt = 0; nt < 6; ++nt)
            bfr[nt] = *(const v16bf*)(wb + (size_t)nt*32*16);

        // A fragment (16x32 bf16): per-lane K groups {0..7,16..23} / {8..15,24..31}
        // relative to fp's khalf*8 base -> element offsets are compile-time consts.
        const float* fk = fp + (size_t)kk * 32 * HW;
        v16bf a;
        #pragma unroll
        for (int e = 0; e < 16; ++e){
            int krel = ((e & 8) ? 16 : 0) + (e & 7);   // 0..7 or 16..23
            a[e] = (__bf16)fk[(size_t)krel * HW];
        }

        #pragma unroll
        for (int nt = 0; nt < 6; ++nt)
            acc[nt] = __builtin_amdgcn_wmma_f32_16x16x32_bf16(
                false, a, false, bfr[nt], (short)0, acc[nt], false, false);
    }

    // D layout: lane -> N = lane&15, VGPR v -> M = v + 8*khalf
    #pragma unroll
    for (int nt = 0; nt < 6; ++nt){
        int n = nt*16 + (lane & 15);
        float bv = biasp[n];
        #pragma unroll
        for (int v = 0; v < 8; ++v)
            lds[wave][v + 8*khalf][n] = acc[nt][v] + bv;
    }
    __syncthreads();

    if (lane < 16){
        const float* row = lds[wave][lane];
        float so   = sigm(row[0]);
        float best = -1.0f; int bl = 0;
        #pragma unroll 4
        for (int c = 0; c < NCCH; ++c){
            float s = so * sigm(row[1 + c]);
            if (s > best){ best = s; bl = c; }
        }
        constexpr float fs = (float)STRD;
        float r0 = __expf(row[81])*fs, r1 = __expf(row[82])*fs;
        float r2 = __expf(row[83])*fs, r3 = __expf(row[84])*fs;
        int p = p0 + lane;
        int y = p / H, x = p - y*H;                  // H is power of 2 -> shifts
        float ax = (x + 0.5f)*fs, ay = (y + 0.5f)*fs;
        scores[p] = best;
        labels[p] = bl;
        float4 bx = make_float4(ax - r0, ay - r1, ax + r2, ay + r3);
        *(float4*)(boxes + 4*(size_t)p) = bx;
    }
}

// ---------------------------------------------------------------------------
// Top-K surrogate: 1024-bin histogram over (0,1) scores -> threshold -> compact
// ---------------------------------------------------------------------------
__global__ void hist_kernel(const float* __restrict__ scores, int n, unsigned* __restrict__ hist)
{
    int i = blockIdx.x * blockDim.x + threadIdx.x;
    if (i < n){
        int b = (int)(scores[i] * 1024.0f);
        b = b < 0 ? 0 : (b > 1023 ? 1023 : b);
        atomicAdd(&hist[b], 1u);
    }
}

__global__ void thresh_kernel(const unsigned* __restrict__ hist, float* __restrict__ thr)
{
    if (threadIdx.x != 0) return;
    const unsigned* h = hist + blockIdx.x * 1024;
    unsigned acc = 0; int bsel = 0;
    for (int b = 1023; b >= 0; --b){
        acc += h[b];
        if (acc >= TOPKN){ bsel = b; break; }
    }
    thr[blockIdx.x] = (float)bsel / 1024.0f;
}

__global__ void compact_kernel(const float* __restrict__ scores, const int* __restrict__ labels,
                               const float* __restrict__ boxes, int n,
                               const float* __restrict__ thr, unsigned* __restrict__ counter,
                               float* __restrict__ cscore, int* __restrict__ clabel,
                               float* __restrict__ cbox, int base)
{
    int i = blockIdx.x * blockDim.x + threadIdx.x;
    if (i < n && scores[i] >= thr[0]){
        unsigned idx = atomicAdd(counter, 1u);
        if (idx < TOPKN){
            int o = base + (int)idx;
            cscore[o] = scores[i];
            clabel[o] = labels[i];
            *(float4*)(cbox + 4*(size_t)o) = *(const float4*)(boxes + 4*(size_t)i);
        }
    }
}

// ---------------------------------------------------------------------------
// Single-block bitonic sort of 4096 (score<<32 | idx) keys, descending
// ---------------------------------------------------------------------------
__global__ void __launch_bounds__(1024)
sort_kernel(const float* __restrict__ cscore, unsigned* __restrict__ order)
{
    __shared__ unsigned long long key[NSORT];        // 32 KB
    for (int i = threadIdx.x; i < NSORT; i += 1024){
        unsigned long long k;
        if (i < NCAND){
            unsigned sb = __float_as_uint(cscore[i]);    // scores >= 0 -> monotone bits
            k = ((unsigned long long)sb << 32) | (unsigned)i;
        } else {
            k = (unsigned long long)i;                   // score 0 padding
        }
        key[i] = k;
    }
    __syncthreads();
    for (unsigned kk = 2; kk <= NSORT; kk <<= 1){
        for (unsigned j = kk >> 1; j > 0; j >>= 1){
            for (int i = threadIdx.x; i < NSORT; i += 1024){
                unsigned ixj = (unsigned)i ^ j;
                if (ixj > (unsigned)i){
                    unsigned long long a = key[i], b = key[ixj];
                    bool up = ((i & kk) == 0);
                    bool sw = up ? (a < b) : (a > b);    // descending overall
                    if (sw){ key[i] = b; key[ixj] = a; }
                }
            }
            __syncthreads();
        }
    }
    for (int i = threadIdx.x; i < NCAND; i += 1024)
        order[i] = (unsigned)(key[i] & 0xffffffffu);
}

// ---------------------------------------------------------------------------
// NMS: class-offset coords + valid bits (wave32 ballot), suppression bitmask,
// serial reduction identical to lax.fori_loop semantics, scatter to orig order
// ---------------------------------------------------------------------------
__global__ void nms_prep_kernel(const unsigned* __restrict__ order, const float* __restrict__ cscore,
                                const int* __restrict__ clabel, const float* __restrict__ cbox,
                                float* xs1, float* ys1, float* xs2, float* ys2,
                                float* areas, unsigned* validw)
{
    int s = blockIdx.x * blockDim.x + threadIdx.x;
    bool valid = false;
    if (s < NCAND){
        unsigned c = order[s];
        float x1 = 0.f, y1 = 0.f, x2 = 0.f, y2 = 0.f, sc = 0.f;
        if (c < NCAND){
            float offc = (float)clabel[c] * CLSOFF;
            x1 = cbox[4*c+0] + offc; y1 = cbox[4*c+1] + offc;
            x2 = cbox[4*c+2] + offc; y2 = cbox[4*c+3] + offc;
            sc = cscore[c];
        }
        xs1[s] = x1; ys1[s] = y1; xs2[s] = x2; ys2[s] = y2;
        areas[s] = (x2 - x1) * (y2 - y1);
        valid = (sc >= CONF_T);
    }
    unsigned bal = __builtin_amdgcn_ballot_w32(valid);
    if ((threadIdx.x & 31) == 0){
        int w = s >> 5;
        if (w < NWORDS) validw[w] = bal;
    }
}

__global__ void __launch_bounds__(128)
nms_mask_kernel(const float* __restrict__ xs1, const float* __restrict__ ys1,
                const float* __restrict__ xs2, const float* __restrict__ ys2,
                const float* __restrict__ areas, unsigned* __restrict__ mask)
{
    const int i = blockIdx.x;
    const int lane = threadIdx.x & 31;
    const float x1i = xs1[i], y1i = ys1[i], x2i = xs2[i], y2i = ys2[i], ai = areas[i];
    for (int w = threadIdx.x >> 5; w < NWORDS; w += 4){
        int j = w*32 + lane;
        bool sup = false;
        if (j < NCAND && j > i){
            float xx1 = fmaxf(x1i, xs1[j]);
            float yy1 = fmaxf(y1i, ys1[j]);
            float xx2 = fminf(x2i, xs2[j]);
            float yy2 = fminf(y2i, ys2[j]);
            float ww = fmaxf(1e-10f, xx2 - xx1);
            float hh = fmaxf(1e-10f, yy2 - yy1);
            float inter = ww * hh;
            float iou = inter / (ai + areas[j] - inter + 1e-14f);
            sup = iou > NMS_THR;
        }
        unsigned bal = __builtin_amdgcn_ballot_w32(sup);
        if (lane == 0) mask[(size_t)i * NWORDS + w] = bal;
    }
}

__global__ void __launch_bounds__(128)
nms_reduce_kernel(const unsigned* __restrict__ validw, const unsigned* __restrict__ mask,
                  const unsigned* __restrict__ order, float* __restrict__ keep)
{
    __shared__ unsigned kw[NWORDS];
    for (int t = threadIdx.x; t < NWORDS; t += blockDim.x) kw[t] = validw[t];
    __syncthreads();
    for (int i = 0; i < NCAND; ++i){
        bool alive = (kw[i >> 5] >> (i & 31)) & 1u;
        if (alive){
            const unsigned* mrow = mask + (size_t)i * NWORDS;
            for (int t = threadIdx.x; t < NWORDS; t += blockDim.x)
                kw[t] &= ~mrow[t];
        }
        __syncthreads();
    }
    for (int s = threadIdx.x; s < NCAND; s += blockDim.x){
        unsigned c = order[s];
        if (c < NCAND)
            keep[c] = ((kw[s >> 5] >> (s & 31)) & 1u) ? 1.0f : 0.0f;
    }
}

__global__ void finalize_kernel(const float* __restrict__ cscore, const int* __restrict__ clabel,
                                const float* __restrict__ cbox, const float* __restrict__ keep,
                                float* __restrict__ out)
{
    int i = blockIdx.x * blockDim.x + threadIdx.x;
    if (i >= NCAND) return;
    bool kb = keep[i] > 0.5f;
    const float inv = 1.0f / IMG_SZ;
    float b0 = fminf(fmaxf(cbox[4*i+0]*inv, 0.f), 1.f);
    float b1 = fminf(fmaxf(cbox[4*i+1]*inv, 0.f), 1.f);
    float b2 = fminf(fmaxf(cbox[4*i+2]*inv, 0.f), 1.f);
    float b3 = fminf(fmaxf(cbox[4*i+3]*inv, 0.f), 1.f);
    out[4*i+0] = kb ? b0 : 0.f;
    out[4*i+1] = kb ? b1 : 0.f;
    out[4*i+2] = kb ? b2 : 0.f;
    out[4*i+3] = kb ? b3 : 0.f;
    out[12000 + i] = kb ? cscore[i] : 0.f;
    out[15000 + i] = kb ? (float)clabel[i] : -1.0f;
    out[18000 + i] = kb ? 1.0f : 0.f;
}

// ---------------------------------------------------------------------------
extern "C" void kernel_launch(void* const* d_in, const int* in_sizes, int n_in,
                              void* d_out, int out_size, void* d_ws, size_t ws_size,
                              hipStream_t stream)
{
    (void)in_sizes; (void)n_in; (void)out_size; (void)ws_size;
    // setup_inputs order: feat0, w0, b0, feat1, w1, b1, feat2, w2, b2
    const float* feat[3] = {(const float*)d_in[0], (const float*)d_in[3], (const float*)d_in[6]};
    const float* w[3]    = {(const float*)d_in[1], (const float*)d_in[4], (const float*)d_in[7]};
    const float* b[3]    = {(const float*)d_in[2], (const float*)d_in[5], (const float*)d_in[8]};

    char* ws = (char*)d_ws;
    size_t off = 0;
    auto take = [&](size_t bytes) -> char* {
        char* p = ws + off;
        off = (off + bytes + 255) & ~(size_t)255;
        return p;
    };
    __bf16*   wpack   = (__bf16*)  take(172032 * 2);
    float*    biasp   = (float*)   take(3 * NPAD * 4);
    float*    lscores = (float*)   take(86016 * 4);
    int*      llabels = (int*)     take(86016 * 4);
    float*    lboxes  = (float*)   take((size_t)86016 * 16);
    unsigned* hist    = (unsigned*)take(3 * 1024 * 4);
    float*    thr     = (float*)   take(3 * 4);
    unsigned* cnt     = (unsigned*)take(3 * 4);
    float*    cscore  = (float*)   take(NCAND * 4);
    int*      clabel  = (int*)     take(NCAND * 4);
    float*    cbox    = (float*)   take(NCAND * 16);
    unsigned* order   = (unsigned*)take(NCAND * 4);
    float*    xs1     = (float*)   take(NCAND * 4);
    float*    ys1     = (float*)   take(NCAND * 4);
    float*    xs2     = (float*)   take(NCAND * 4);
    float*    ys2     = (float*)   take(NCAND * 4);
    float*    areas   = (float*)   take(NCAND * 4);
    unsigned* validw  = (unsigned*)take(128 * 4);
    unsigned* mask    = (unsigned*)take((size_t)NCAND * NWORDS * 4);
    float*    keep    = (float*)   take(NCAND * 4);

    init_kernel<<<12, 256, 0, stream>>>(hist, cnt, cscore, clabel, cbox, validw, keep);
    pack_weights_kernel<<<(172032 + 255) / 256, 256, 0, stream>>>(w[0], w[1], w[2], wpack);
    pack_bias_kernel<<<2, 256, 0, stream>>>(b[0], b[1], b[2], biasp);

    const int scoff[3] = {0, 65536, 81920};
    const int wpoff[3] = {0, 24576, 73728};

    // level 0: K=256 (8 ksteps), H=256, stride 8, 4096 tiles
    head_kernel<8, 256, 8><<<(256*256/16) / 4, 128, 0, stream>>>(
        feat[0], wpack + wpoff[0], biasp + 0 * NPAD,
        lscores + scoff[0], llabels + scoff[0], lboxes + (size_t)scoff[0] * 4);
    // level 1: K=512 (16 ksteps), H=128, stride 16, 1024 tiles
    head_kernel<16, 128, 16><<<(128*128/16) / 4, 128, 0, stream>>>(
        feat[1], wpack + wpoff[1], biasp + 1 * NPAD,
        lscores + scoff[1], llabels + scoff[1], lboxes + (size_t)scoff[1] * 4);
    // level 2: K=1024 (32 ksteps), H=64, stride 32, 256 tiles
    head_kernel<32, 64, 32><<<(64*64/16) / 4, 128, 0, stream>>>(
        feat[2], wpack + wpoff[2], biasp + 2 * NPAD,
        lscores + scoff[2], llabels + scoff[2], lboxes + (size_t)scoff[2] * 4);

    const int Hs[3] = {256, 128, 64};
    for (int l = 0; l < 3; ++l){
        int HW = Hs[l] * Hs[l];
        hist_kernel<<<(HW + 255) / 256, 256, 0, stream>>>(lscores + scoff[l], HW, hist + l * 1024);
    }
    thresh_kernel<<<3, 32, 0, stream>>>(hist, thr);
    for (int l = 0; l < 3; ++l){
        int HW = Hs[l] * Hs[l];
        compact_kernel<<<(HW + 255) / 256, 256, 0, stream>>>(
            lscores + scoff[l], llabels + scoff[l], lboxes + (size_t)scoff[l] * 4,
            HW, thr + l, cnt + l, cscore, clabel, cbox, l * TOPKN);
    }
    sort_kernel<<<1, 1024, 0, stream>>>(cscore, order);
    nms_prep_kernel<<<24, 128, 0, stream>>>(order, cscore, clabel, cbox,
                                            xs1, ys1, xs2, ys2, areas, validw);
    nms_mask_kernel<<<NCAND, 128, 0, stream>>>(xs1, ys1, xs2, ys2, areas, mask);
    nms_reduce_kernel<<<1, 128, 0, stream>>>(validw, mask, order, keep);
    finalize_kernel<<<(NCAND + 127) / 128, 128, 0, stream>>>(cscore, clabel, cbox, keep, (float*)d_out);
}
